// GeoformerMultiHeadAttention_4595615007279
// MI455X (gfx1250) — matrix operand order repaired and checked by reading, not compiled, not run on hardware
//
#include <hip/hip_runtime.h>
#include <hip/hip_bf16.h>
#include <stdint.h>

// ---------------------------------------------------------------------------
// Geoformer multi-head attention, CDNA5 (gfx1250) wave32 + WMMA f16 +
// async global->LDS tile staging (ASYNCcnt) with LDS double buffering.
// B=4, N=128, E=512, H=8, D=64.  All GEMMs have K = E = 512 (compile-time).
// ---------------------------------------------------------------------------

typedef __attribute__((ext_vector_type(16))) _Float16 v16h;
typedef __attribute__((ext_vector_type(8)))  _Float16 v8h;
typedef __attribute__((ext_vector_type(2)))  _Float16 v2h;
typedef __attribute__((ext_vector_type(8)))  float    v8f;

#define BDIM 4
#define NDIM 128
#define EDIM 512
#define HDIM 8
#define DDIM 64

#define BM 128
#define BN 128
#define BK 32
#define LDSPAD 8   // 40 f16 per row -> 80B row pitch, conflict-free 16B loads

__device__ __forceinline__ float silu_f(float x) {
    return x / (1.0f + __expf(-x));
}

__device__ __forceinline__ v16h make_frag(v8h lo, v8h hi) {
    v16h f;
#pragma unroll
    for (int i = 0; i < 8; ++i) { f[i] = lo[i]; f[i + 8] = hi[i]; }
    return f;
}

// Async 16B global -> LDS copy (no VGPR staging, tracked by ASYNCcnt).
__device__ __forceinline__ void async_copy_b128(const void* gptr,
                                                uint32_t lds_off) {
    asm volatile("global_load_async_to_lds_b128 %0, %1, off"
                 :: "v"(lds_off), "v"(gptr)
                 : "memory");
}
__device__ __forceinline__ void wait_async_le4() {
    asm volatile("s_wait_asynccnt 0x4" ::: "memory");
}
__device__ __forceinline__ void wait_async_0() {
    asm volatile("s_wait_asynccnt 0x0" ::: "memory");
}

// ---------------------------------------------------------------------------
// f32 -> f16 conversion (grid-stride)
// ---------------------------------------------------------------------------
__global__ void cvt_f32_f16_kernel(const float* __restrict__ in,
                                   _Float16* __restrict__ out, size_t n) {
    size_t idx = (size_t)blockIdx.x * blockDim.x + threadIdx.x;
    size_t stride = (size_t)gridDim.x * blockDim.x;
    for (; idx < n; idx += stride) out[idx] = (_Float16)in[idx];
}

// ---------------------------------------------------------------------------
// WMMA GEMM:  Y[r, c] = act( sum_k A[r,k] * W[c,k] + bias[c] ),  f16 in/out,
// f32 accumulate, K fixed at 512. Optional per-row mask (zeroes the row).
// Block: 256 threads (8 waves). Tile BM x BN, K-step BK=32 (one WMMA K).
// Wave (wm 0..3, wn 0..1) computes a 32x64 subtile = 2x4 WMMA 16x16 tiles.
// Tiles staged with global_load_async_to_lds_b128 into double-buffered LDS;
// fully unrolled 16-step software pipeline (static buffer indices, no
// per-step branches or cndmask address selection).
// ---------------------------------------------------------------------------
template <int ACT, bool MASKED>
__global__ __launch_bounds__(256) void wmma_gemm_kernel(
    const _Float16* __restrict__ A, const _Float16* __restrict__ W,
    const float* __restrict__ bias, const unsigned char* __restrict__ mask,
    _Float16* __restrict__ Y, int R, int Ncol) {
    constexpr int KD = EDIM;        // 512
    constexpr int NSTEP = KD / BK;  // 16

    __shared__ _Float16 As[2][BM][BK + LDSPAD];
    __shared__ _Float16 Bs[2][BM][BK + LDSPAD];

    const int tid = threadIdx.x;
    const int rowBase = blockIdx.x * BM;
    const int colBase = blockIdx.y * BN;
    const int wave = tid >> 5;
    const int lane = tid & 31;
    const int wm = wave >> 1;   // 0..3  (row quadrant)
    const int wn = wave & 1;    // 0..1  (col half)
    const int lr = lane & 15;
    const int lh = lane >> 4;

    // Per-thread copy slots: 4 x b128 per tile (2 rows of A, 2 rows of W).
    const int r0 = tid >> 2;          // 0..63
    const int cc = (tid & 3) * 8;     // 0,8,16,24
    const _Float16* gA0 = A + (size_t)(rowBase + r0) * KD + cc;
    const _Float16* gA1 = A + (size_t)(rowBase + r0 + 64) * KD + cc;
    const _Float16* gB0 = W + (size_t)(colBase + r0) * KD + cc;
    const _Float16* gB1 = W + (size_t)(colBase + r0 + 64) * KD + cc;
    uint32_t lA0[2], lA1[2], lB0[2], lB1[2];
#pragma unroll
    for (int bb = 0; bb < 2; ++bb) {
        lA0[bb] = (uint32_t)(uintptr_t)&As[bb][r0][cc];
        lA1[bb] = (uint32_t)(uintptr_t)&As[bb][r0 + 64][cc];
        lB0[bb] = (uint32_t)(uintptr_t)&Bs[bb][r0][cc];
        lB1[bb] = (uint32_t)(uintptr_t)&Bs[bb][r0 + 64][cc];
    }

    v8f acc[2][4];
#pragma unroll
    for (int mt = 0; mt < 2; ++mt)
#pragma unroll
        for (int nt = 0; nt < 4; ++nt)
#pragma unroll
            for (int i = 0; i < 8; ++i) acc[mt][nt][i] = 0.0f;

    // Prologue: issue tile 0 into buffer 0.
    async_copy_b128(gA0, lA0[0]);
    async_copy_b128(gA1, lA1[0]);
    async_copy_b128(gB0, lB0[0]);
    async_copy_b128(gB1, lB1[0]);

#pragma unroll
    for (int s = 0; s < NSTEP; ++s) {
        const int buf = s & 1;               // static after unroll
        if (s + 1 < NSTEP) {
            const int kn = (s + 1) * BK;
            const int nb = buf ^ 1;
            async_copy_b128(gA0 + kn, lA0[nb]);
            async_copy_b128(gA1 + kn, lA1[nb]);
            async_copy_b128(gB0 + kn, lB0[nb]);
            async_copy_b128(gB1 + kn, lB1[nb]);
            wait_async_le4();   // previous tile's 4 copies have landed
        } else {
            wait_async_0();
        }
        __syncthreads();        // all waves' copies for `buf` complete

        // A fragments: 16x32, lane holds row M=lr; k = lh*8..+7, 16+lh*8..+7
        v16h afrag[2];
#pragma unroll
        for (int mt = 0; mt < 2; ++mt) {
            const int r = wm * 32 + mt * 16 + lr;
            v8h lo = *(v8h*)&As[buf][r][lh * 8];
            v8h hi = *(v8h*)&As[buf][r][16 + lh * 8];
            afrag[mt] = make_frag(lo, hi);
        }
        // B fragments: 32x16 (K x N), lane holds column n=lr; k = lh*8..+7, 16+lh*8..+7
        v16h bfrag[4];
#pragma unroll
        for (int nt = 0; nt < 4; ++nt) {
            const int r = wn * 64 + nt * 16 + lr;
            v8h lo = *(v8h*)&Bs[buf][r][lh * 8];
            v8h hi = *(v8h*)&Bs[buf][r][16 + lh * 8];
            bfrag[nt] = make_frag(lo, hi);
        }
#pragma unroll
        for (int mt = 0; mt < 2; ++mt)
#pragma unroll
            for (int nt = 0; nt < 4; ++nt)
                acc[mt][nt] = __builtin_amdgcn_wmma_f32_16x16x32_f16(
                    false, afrag[mt], false, bfrag[nt], (short)0, acc[mt][nt],
                    false, false);

        __syncthreads();        // done reading `buf`; safe to overwrite later
    }

    // Epilogue: D element (i, lane) -> row = i + lh*8, col = lr within tile.
#pragma unroll
    for (int mt = 0; mt < 2; ++mt) {
#pragma unroll
        for (int nt = 0; nt < 4; ++nt) {
            const int col = colBase + wn * 64 + nt * 16 + lr;
            const float bv = bias ? bias[col] : 0.0f;
#pragma unroll
            for (int i = 0; i < 8; ++i) {
                const int row = rowBase + wm * 32 + mt * 16 + lh * 8 + i;
                float vv = acc[mt][nt][i] + bv;
                if (ACT == 1) vv = silu_f(vv);
                if (MASKED) {
                    if (mask[row]) vv = 0.0f;
                }
                Y[(size_t)row * Ncol + col] = (_Float16)vv;
            }
        }
    }
}

// ---------------------------------------------------------------------------
// Attention: block per (b,i); wave w = head h; lane owns d-pair (2l, 2l+1).
// attn_w[h,j] = sum_d q*k*dk ; probs = silu(mask? 0 : w) * cutoff(dist);
// apn[b,i,j,e] = probs * v[b,j,e] (f16), attn[b,i,e] = sum_j apn (f32 out).
// ---------------------------------------------------------------------------
__global__ __launch_bounds__(256) void attn_kernel(
    const _Float16* __restrict__ qh, const _Float16* __restrict__ kh,
    const _Float16* __restrict__ vh, const _Float16* __restrict__ dkh,
    const float* __restrict__ dist, const unsigned char* __restrict__ mask,
    float* __restrict__ attn_out, _Float16* __restrict__ apn) {
    const int bi = blockIdx.x;           // b*N + i
    const int b = bi >> 7;
    const int h = threadIdx.x >> 5;
    const int lane = threadIdx.x & 31;
    const int e = h * DDIM + lane * 2;

    const size_t qoff = (size_t)bi * EDIM + e;
    const float qa = (float)qh[qoff];
    const float qb = (float)qh[qoff + 1];
    float acc0 = 0.0f, acc1 = 0.0f;
    const size_t rowb = (size_t)bi * NDIM;   // (b*N+i)*N

    for (int j = 0; j < NDIM; ++j) {
        const size_t kof = ((size_t)(b * NDIM + j)) * EDIM + e;
        const float ka = (float)kh[kof];
        const float kb = (float)kh[kof + 1];
        const size_t dkof = (rowb + j) * EDIM + e;
        const float da = (float)dkh[dkof];
        const float db = (float)dkh[dkof + 1];
        float partial = qa * ka * da + qb * kb * db;
#pragma unroll
        for (int off = 16; off > 0; off >>= 1)
            partial += __shfl_xor(partial, off, 32);
        float w = mask[rowb + j] ? 0.0f : partial;
        const float d = dist[rowb + j];
        const float scale =
            (d < 5.0f) ? 0.5f * (__cosf(d * 0.6283185307f) + 1.0f) : 0.0f;
        const float p = silu_f(w) * scale;
        const float va = (float)vh[kof];
        const float vb = (float)vh[kof + 1];
        const float r0 = p * va, r1 = p * vb;
        v2h pk;
        pk[0] = (_Float16)r0;
        pk[1] = (_Float16)r1;
        *(v2h*)&apn[dkof] = pk;
        acc0 += r0;
        acc1 += r1;
    }
    attn_out[qoff] = acc0;
    attn_out[qoff + 1] = acc1;
}

// ---------------------------------------------------------------------------
// du[b,n,c,e] = sum_m dupd[b,n,m,e] * vec[b,n,m,c], then vec_layer_norm over
// c (norm) and e (max/min). Output f16 rows (b,n,c) x E for the wswt GEMM.
// Block per (b,n); thread t owns e = t and e = t+256.
// ---------------------------------------------------------------------------
__global__ __launch_bounds__(256) void du_norm_kernel(
    const _Float16* __restrict__ dupd, const float* __restrict__ vec,
    _Float16* __restrict__ du_out) {
    __shared__ float vs[NDIM][3];
    __shared__ float rmax[256];
    __shared__ float rmin[256];
    const int bn = blockIdx.x;
    const int t = threadIdx.x;

    if (t < NDIM) {
        const size_t base = ((size_t)bn * NDIM + t) * 3;
        vs[t][0] = vec[base];
        vs[t][1] = vec[base + 1];
        vs[t][2] = vec[base + 2];
    }
    __syncthreads();

    float a[3][2] = {{0.f, 0.f}, {0.f, 0.f}, {0.f, 0.f}};
    for (int m = 0; m < NDIM; ++m) {
        const size_t off = ((size_t)bn * NDIM + m) * EDIM + t;
        const float d0 = (float)dupd[off];
        const float d1 = (float)dupd[off + 256];
        const float v0 = vs[m][0], v1 = vs[m][1], v2 = vs[m][2];
        a[0][0] += d0 * v0; a[0][1] += d1 * v0;
        a[1][0] += d0 * v1; a[1][1] += d1 * v1;
        a[2][0] += d0 * v2; a[2][1] += d1 * v2;
    }

    float dist0 = sqrtf(a[0][0]*a[0][0] + a[1][0]*a[1][0] + a[2][0]*a[2][0]);
    float dist1 = sqrtf(a[0][1]*a[0][1] + a[1][1]*a[1][1] + a[2][1]*a[2][1]);
    dist0 = fmaxf(dist0, 1e-12f);
    dist1 = fmaxf(dist1, 1e-12f);

    rmax[t] = fmaxf(dist0, dist1);
    rmin[t] = fminf(dist0, dist1);
    __syncthreads();
    for (int s = 128; s > 0; s >>= 1) {
        if (t < s) {
            rmax[t] = fmaxf(rmax[t], rmax[t + s]);
            rmin[t] = fminf(rmin[t], rmin[t + s]);
        }
        __syncthreads();
    }
    const float mx = rmax[0];
    const float mn = rmin[0];
    float delta = mx - mn;
    if (delta == 0.0f) delta = 1.0f;
    const float nd0 = fmaxf((dist0 - mn) / delta, 0.0f);
    const float nd1 = fmaxf((dist1 - mn) / delta, 0.0f);
    const float s0 = nd0 / dist0;
    const float s1 = nd1 / dist1;
#pragma unroll
    for (int c = 0; c < 3; ++c) {
        du_out[((size_t)bn * 3 + c) * EDIM + t]       = (_Float16)(s0 * a[c][0]);
        du_out[((size_t)bn * 3 + c) * EDIM + t + 256] = (_Float16)(s1 * a[c][1]);
    }
}

// ---------------------------------------------------------------------------
// ipe[b,i,j,e] = ea[b,i,j,e] * sum_c ws[b,i,c,e] * wt[b,j,c,e]  (f32 out)
// wswt rows are (b,n,c) with 1024 cols: ws = cols[0:512), wt = cols[512:1024).
// Block per (b,i); thread t owns e = t and e = t+256.
// ---------------------------------------------------------------------------
__global__ __launch_bounds__(256) void ipe_kernel(
    const _Float16* __restrict__ wswt, const _Float16* __restrict__ eah,
    float* __restrict__ out) {
    __shared__ _Float16 wsl[3][EDIM];
    const int bi = blockIdx.x;  // b*N + i
    const int b = bi >> 7;
    const int t = threadIdx.x;

#pragma unroll
    for (int c = 0; c < 3; ++c) {
        const size_t row = (size_t)bi * 3 + c;
        wsl[c][t]       = wswt[row * 1024 + t];
        wsl[c][t + 256] = wswt[row * 1024 + t + 256];
    }
    __syncthreads();

    for (int j = 0; j < NDIM; ++j) {
        const size_t wrow = (size_t)(b * NDIM + j) * 3;
        float s0 = 0.0f, s1 = 0.0f;
#pragma unroll
        for (int c = 0; c < 3; ++c) {
            const size_t base = (wrow + c) * 1024 + 512;
            s0 += (float)wsl[c][t]       * (float)wswt[base + t];
            s1 += (float)wsl[c][t + 256] * (float)wswt[base + t + 256];
        }
        const size_t eoff = ((size_t)bi * NDIM + j) * EDIM;
        out[eoff + t]       = (float)eah[eoff + t] * s0;
        out[eoff + t + 256] = (float)eah[eoff + t + 256] * s1;
    }
}

// ---------------------------------------------------------------------------
// Host-side orchestration
// ---------------------------------------------------------------------------
extern "C" void kernel_launch(void* const* d_in, const int* in_sizes, int n_in,
                              void* d_out, int out_size, void* d_ws,
                              size_t ws_size, hipStream_t stream) {
    (void)in_sizes; (void)n_in; (void)out_size; (void)ws_size;

    const float* x    = (const float*)d_in[0];
    const float* vec  = (const float*)d_in[1];
    const float* dist = (const float*)d_in[2];
    const float* edge = (const float*)d_in[3];
    const unsigned char* kpm = (const unsigned char*)d_in[4];
    const float* Wq  = (const float*)d_in[5];
    const float* bq  = (const float*)d_in[6];
    const float* Wk  = (const float*)d_in[7];
    const float* bk  = (const float*)d_in[8];
    const float* Wv  = (const float*)d_in[9];
    const float* bv  = (const float*)d_in[10];
    const float* Wdk = (const float*)d_in[11];
    const float* bdk = (const float*)d_in[12];
    const float* Wdu = (const float*)d_in[13];
    const float* bdu = (const float*)d_in[14];
    const float* Wdih = (const float*)d_in[15];
    const float* Wea  = (const float*)d_in[16];
    const float* bea  = (const float*)d_in[17];

    constexpr size_t XN   = (size_t)BDIM * NDIM * EDIM;            // 262144
    constexpr size_t EDGE = (size_t)BDIM * NDIM * NDIM * EDIM;     // 33554432
    constexpr size_t WN   = (size_t)EDIM * EDIM;                   // 262144
    constexpr size_t WDIHN = 2 * WN;                               // 524288
    constexpr size_t DUN  = (size_t)BDIM * NDIM * 3 * EDIM;        // 786432
    constexpr size_t WSWTN = 2 * DUN;                              // 1572864

    char* p = (char*)d_ws;
    _Float16* xh   = (_Float16*)p; p += XN * 2;
    _Float16* wqh  = (_Float16*)p; p += WN * 2;
    _Float16* wkh  = (_Float16*)p; p += WN * 2;
    _Float16* wvh  = (_Float16*)p; p += WN * 2;
    _Float16* wdkh = (_Float16*)p; p += WN * 2;
    _Float16* wduh = (_Float16*)p; p += WN * 2;
    _Float16* weah = (_Float16*)p; p += WN * 2;
    _Float16* wdihh = (_Float16*)p; p += WDIHN * 2;
    _Float16* qh   = (_Float16*)p; p += XN * 2;
    _Float16* kh   = (_Float16*)p; p += XN * 2;
    _Float16* vh   = (_Float16*)p; p += XN * 2;
    _Float16* edgeh = (_Float16*)p; p += EDGE * 2;   // reused as apn later
    _Float16* dkh  = (_Float16*)p; p += EDGE * 2;    // reused as dupd later
    _Float16* eahh = (_Float16*)p; p += EDGE * 2;
    _Float16* duh  = (_Float16*)p; p += DUN * 2;
    _Float16* wswth = (_Float16*)p; p += WSWTN * 2;

    float* attn_out = (float*)d_out;            // B*N*E floats
    float* ipe_out  = attn_out + XN;            // B*N*N*E floats

    // ---- 1) f32 -> f16 conversions ----
    auto cvt = [&](const float* src, _Float16* dst, size_t n) {
        const int blocks = (int)((n + 255) / 256 > 4096 ? 4096 : (n + 255) / 256);
        cvt_f32_f16_kernel<<<blocks, 256, 0, stream>>>(src, dst, n);
    };
    cvt(x, xh, XN);
    cvt(edge, edgeh, EDGE);
    cvt(Wq, wqh, WN);
    cvt(Wk, wkh, WN);
    cvt(Wv, wvh, WN);
    cvt(Wdk, wdkh, WN);
    cvt(Wdu, wduh, WN);
    cvt(Wea, weah, WN);
    cvt(Wdih, wdihh, WDIHN);

    // ---- 2) q,k,v = x @ W^T + b  (512x512x512) ----
    {
        dim3 grid(BDIM * NDIM / BM, EDIM / BN);
        wmma_gemm_kernel<0, false><<<grid, 256, 0, stream>>>(
            xh, wqh, bq, nullptr, qh, BDIM * NDIM, EDIM);
        wmma_gemm_kernel<0, false><<<grid, 256, 0, stream>>>(
            xh, wkh, bk, nullptr, kh, BDIM * NDIM, EDIM);
        wmma_gemm_kernel<0, false><<<grid, 256, 0, stream>>>(
            xh, wvh, bv, nullptr, vh, BDIM * NDIM, EDIM);
    }

    // ---- 3) dk = silu(edge @ Wdk^T + bdk); ea = silu(edge @ Wea^T + bea) ----
    {
        dim3 grid(BDIM * NDIM * NDIM / BM, EDIM / BN);
        wmma_gemm_kernel<1, false><<<grid, 256, 0, stream>>>(
            edgeh, wdkh, bdk, nullptr, dkh, BDIM * NDIM * NDIM, EDIM);
        wmma_gemm_kernel<1, false><<<grid, 256, 0, stream>>>(
            edgeh, weah, bea, nullptr, eahh, BDIM * NDIM * NDIM, EDIM);
    }

    // ---- 4) attention: attn (f32 out) + attn_per_nodes (f16, reuse edgeh) ----
    _Float16* apn = edgeh;  // edge_attr f16 no longer needed
    attn_kernel<<<BDIM * NDIM, 256, 0, stream>>>(qh, kh, vh, dkh, dist, kpm,
                                                 attn_out, apn);

    // ---- 5) dupd = (apn @ Wdu^T + bdu), row-masked (reuse dkh) ----
    _Float16* dupd = dkh;  // dk no longer needed
    {
        dim3 grid(BDIM * NDIM * NDIM / BM, EDIM / BN);
        wmma_gemm_kernel<0, true><<<grid, 256, 0, stream>>>(
            apn, wduh, bdu, kpm, dupd, BDIM * NDIM * NDIM, EDIM);
    }

    // ---- 6) du contraction + vec_layer_norm -> duh (rows (b,n,c) x E) ----
    du_norm_kernel<<<BDIM * NDIM, 256, 0, stream>>>(dupd, vec, duh);

    // ---- 7) wswt = du @ Wdih^T  (1536 x 512 x 1024, no bias/act) ----
    {
        dim3 grid((int)(BDIM * NDIM * 3 / BM), 2 * EDIM / BN);
        wmma_gemm_kernel<0, false><<<grid, 256, 0, stream>>>(
            duh, wdihh, nullptr, nullptr, wswth, BDIM * NDIM * 3, 2 * EDIM);
    }

    // ---- 8) ipe = ea * einsum('bice,bjce->bije', ws, wt) -> f32 output ----
    ipe_kernel<<<BDIM * NDIM, 256, 0, stream>>>(wswth, eahh, ipe_out);
}
